// MultiHeadAttention_6975026888881
// MI455X (gfx1250) — matrix operand order repaired and checked by reading, not compiled, not run on hardware
//
#include <hip/hip_runtime.h>

// ---------------------------------------------------------------------------
// MI455X (gfx1250, wave32) fused Multi-Head Attention
//   out  = LayerNorm( (softmax(QWq (KWk)^T / sqrt(64)) VWv) Wo + b_o + q )
//   attn = softmax scores, head-major [(H*B), Lq, Lk]
// GEMMs: v_wmma_f32_16x16x32_f16, tiles staged by the Tensor Data Mover
// (tensor_load_to_lds, TENSORcnt) with double-buffered LDS.
// ---------------------------------------------------------------------------

typedef __attribute__((ext_vector_type(16))) _Float16 v16h;
typedef __attribute__((ext_vector_type(4)))  _Float16 v4h;
typedef __attribute__((ext_vector_type(8)))  float    v8f;
typedef __attribute__((ext_vector_type(4)))  float    v4f;
typedef __attribute__((ext_vector_type(4)))  unsigned v4u;
typedef __attribute__((ext_vector_type(8)))  int      v8i;
typedef __attribute__((ext_vector_type(4)))  int      v4i;

static constexpr int BATCH = 8;
static constexpr int NH    = 16;
static constexpr int SEQ   = 1024;   // Lq == Lk
static constexpr int DM    = 1024;   // d_model
static constexpr int DK    = 64;     // per-head dim

// ---------------------------------------------------------------------------
// WMMA fragment helpers (wave32 layouts per CDNA5 ISA 7.12.2)
// ---------------------------------------------------------------------------

// A fragment, 16x32 f16 (M x K), row-major source with leading dim `ld`.
// lane<16 : row=lane,    K in {0..7} U {16..23}
// lane>=16: row=lane-16, K in {8..15} U {24..31}
__device__ inline v16h frag_a_f16(const _Float16* base, int ld, int lane) {
  const int row = lane & 15;
  const int kb  = (lane >> 4) ? 8 : 0;
  const _Float16* p = base + row * ld + kb;
  v16h a;
#pragma unroll
  for (int j = 0; j < 8; ++j) { a[j] = p[j]; a[j + 8] = p[j + 16]; }
  return a;
}

// B fragment, 32x16 f16 (K x N), sourced from a TRANSPOSED (column-major)
// buffer Bt[n][k] with leading dim `ld`; lane<16: K {0..7,16..23}, lane>=16: K {8..15,24..31}.
__device__ inline v16h frag_b_t(const _Float16* baseT, int ld, int lane) {
  const int col = lane & 15;
  const int kb  = (lane >> 4) ? 8 : 0;
  const _Float16* p = baseT + col * ld + kb;
  v16h b;
#pragma unroll
  for (int j = 0; j < 8; ++j) { b[j] = p[j]; b[j + 8] = p[j + 16]; }
  return b;
}

__device__ inline v8f wmma32(v16h a, v16h b, v8f c) {
  return __builtin_amdgcn_wmma_f32_16x16x32_f16(
      /*neg_a=*/false, a, /*neg_b=*/false, b,
      /*c_mod=*/(short)0, c, /*reuse_a=*/false, /*reuse_b=*/false);
}

// ---------------------------------------------------------------------------
// TDM: DMA a (tileRows x 32)-halfword tile (row stride = strideElems halves)
// from global into LDS at byte offset ldsOff, padding 1 DWORD after every
// 64 bytes so the LDS row stride becomes 34 halves (bank-conflict relief).
// D# packing per CDNA5 ISA 8.3/8.4; groups 2-3 zero (2-D tile).
// 6-arg builtin form: (g0, g1, g2, g3, g4(unused), cpol).
// ---------------------------------------------------------------------------
__device__ inline void tdm_load_tile(const void* gsrc, unsigned ldsOff,
                                     unsigned tileRows, unsigned strideElems) {
  const unsigned long long ga = (unsigned long long)(size_t)gsrc;
  v4u g0;
  g0[0] = 1u;                                               // count=1 (valid)
  g0[1] = ldsOff;                                           // lds_addr
  g0[2] = (unsigned)ga;                                     // global_addr lo
  g0[3] = ((unsigned)(ga >> 32) & 0x01FFFFFFu) | (2u << 30);// addr[56:32]|type=2

  const unsigned dim0 = strideElems;   // tensor_dim0 (line length, OOB bound)
  const unsigned dim1 = 1u << 20;      // generous row bound
  const unsigned long long s0 = strideElems;  // tensor_dim0_stride

  v8i g1;
  g1[0] = (int)((1u << 16)      // data_size = 2 bytes
              | (1u << 20)      // pad_enable
              | (3u << 22)      // pad_interval: every 16 DWORDs (64B)
              | (0u << 25));    // pad_amount: 1 DWORD
  g1[1] = (int)((dim0 & 0xFFFFu) << 16);
  g1[2] = (int)(((dim0 >> 16) & 0xFFFFu) | ((dim1 & 0xFFFFu) << 16));
  g1[3] = (int)(((dim1 >> 16) & 0xFFFFu) | (32u << 16));    // tile_dim0 = 32
  g1[4] = (int)(tileRows & 0xFFFFu);                        // tile_dim1
  g1[5] = (int)(unsigned)(s0 & 0xFFFFFFFFu);
  g1[6] = (int)(unsigned)((s0 >> 32) & 0xFFFFu);            // dim1_stride = 0
  g1[7] = 0;

  const v4i zero4 = {};
  const v8i zero8 = {};
  __builtin_amdgcn_tensor_load_to_lds(g0, g1, zero4, zero4, zero8, 0);
}

// ---------------------------------------------------------------------------
// Elementwise fp32 -> f16 convert (q, k, v): 4 elems / thread.
// ---------------------------------------------------------------------------
__global__ __launch_bounds__(256) void convert_f16_kernel(
    const float* __restrict__ in, _Float16* __restrict__ out) {
  const size_t i = (size_t)blockIdx.x * 1024 + (size_t)threadIdx.x * 4;
  const v4f x = *(const v4f*)(in + i);
  v4h o;
#pragma unroll
  for (int j = 0; j < 4; ++j) o[j] = (_Float16)x[j];
  *(v4h*)(out + i) = o;
}

// ---------------------------------------------------------------------------
// W[k][n] fp32 -> Wt[n][k] f16 (32x32 LDS tiles).
// ---------------------------------------------------------------------------
__global__ __launch_bounds__(256) void transpose_f16_kernel(
    const float* __restrict__ w, _Float16* __restrict__ wT) {
  __shared__ float t[32][33];
  const int bx = blockIdx.x & 31, by = blockIdx.x >> 5;
  const int tx = threadIdx.x & 31, ty = threadIdx.x >> 5;
#pragma unroll
  for (int i = 0; i < 4; ++i)
    t[ty + i * 8][tx] = w[(size_t)(by * 32 + ty + i * 8) * 1024 + bx * 32 + tx];
  __syncthreads();
#pragma unroll
  for (int i = 0; i < 4; ++i)
    wT[(size_t)(bx * 32 + ty + i * 8) * 1024 + by * 32 + tx] =
        (_Float16)t[tx][ty + i * 8];
}

// ---------------------------------------------------------------------------
// Projection GEMM: out(8192,1024) = X(8192,1024) * Wt^T + bias, all f16 in.
// 256 threads = 8 waves; block tile 128x64; wave computes a 16x64 strip.
// TDM double-buffered staging: wave0 issues next A/B tiles while all compute.
// MODE 0: f16 head-major qh/kh[b][h][t][64]
// MODE 1: f16 transposed vhT[b][h][d][1024]
// MODE 2: f32 + bias + residual -> proj[row][col]
// ---------------------------------------------------------------------------
template <int MODE>
__global__ __launch_bounds__(256) void gemm_kernel(
    const _Float16* __restrict__ X, const _Float16* __restrict__ WT,
    const float* __restrict__ bias, const float* __restrict__ residual,
    void* __restrict__ outv) {
  constexpr int K = 1024, N = 1024;
  constexpr int LDT = 34;                       // padded LDS row stride
  __shared__ _Float16 sA[2][128 * LDT];
  __shared__ _Float16 sB[2][64 * LDT];

  const int nTiles  = N / 64;
  const int rowBase = (blockIdx.x / nTiles) * 128;
  const int colBase = (blockIdx.x % nTiles) * 64;
  const int tid = threadIdx.x, wave = tid >> 5, lane = tid & 31;

  const _Float16* gA = X  + (size_t)rowBase * K;   // 128 rows, stride K
  const _Float16* gB = WT + (size_t)colBase * K;   // 64 rows,  stride K

  v8f acc[4] = {};

  if (wave == 0) {
    tdm_load_tile(gA, (unsigned)(size_t)&sA[0][0], 128, K);
    tdm_load_tile(gB, (unsigned)(size_t)&sB[0][0], 64, K);
  }

  for (int it = 0; it < K / 32; ++it) {
    __builtin_amdgcn_s_wait_tensorcnt(0);   // wave0: current buffer landed
    __syncthreads();                        // publish to all waves
    const int buf = it & 1;
    if (wave == 0 && it + 1 < K / 32) {     // prefetch next tile pair
      tdm_load_tile(gA + (size_t)(it + 1) * 32,
                    (unsigned)(size_t)&sA[buf ^ 1][0], 128, K);
      tdm_load_tile(gB + (size_t)(it + 1) * 32,
                    (unsigned)(size_t)&sB[buf ^ 1][0], 64, K);
    }
    const v16h a = frag_a_f16(&sA[buf][wave * 16 * LDT], LDT, lane);
#pragma unroll
    for (int t = 0; t < 4; ++t) {
      const v16h bf = frag_b_t(&sB[buf][t * 16 * LDT], LDT, lane);
      acc[t] = wmma32(a, bf, acc[t]);
    }
    __syncthreads();                        // done reading buf before reuse
  }

  // Epilogue: C/D layout -> lane holds col = lane&15, rows r + 8*(lane>>4).
  const int colIn = lane & 15, hi = lane >> 4;
#pragma unroll
  for (int t = 0; t < 4; ++t) {
#pragma unroll
    for (int r = 0; r < 8; ++r) {
      const int row = rowBase + wave * 16 + r + 8 * hi;  // token index
      const int col = colBase + t * 16 + colIn;          // h*64 + d
      float v = acc[t][r] + bias[col];
      if (MODE == 0) {
        const int b = row >> 10, tq = row & 1023, h = col >> 6, d = col & 63;
        ((_Float16*)outv)[(((size_t)(b * NH + h) * SEQ + tq) << 6) + d] = (_Float16)v;
      } else if (MODE == 1) {
        const int b = row >> 10, tq = row & 1023, h = col >> 6, d = col & 63;
        ((_Float16*)outv)[(((size_t)(b * NH + h) * DK + d) << 10) + tq] = (_Float16)v;
      } else {
        v += residual[(size_t)row * N + col];
        ((float*)outv)[(size_t)row * N + col] = v;
      }
    }
  }
}

// ---------------------------------------------------------------------------
// Fused attention: one block per (b, h, 16-query strip).
// S = Q K^T / 8 (WMMA) -> LDS; softmax; write attn fp32; P V (WMMA, split-K).
// ---------------------------------------------------------------------------
__global__ __launch_bounds__(256) void attn_kernel(
    const _Float16* __restrict__ qh, const _Float16* __restrict__ kh,
    const _Float16* __restrict__ vhT, const unsigned char* __restrict__ mask,
    float* __restrict__ attn, _Float16* __restrict__ headout) {
  __shared__ float sS[16 * 1024];    // scores -> exp -> probabilities
  __shared__ float sRed[16 * 16];
  __shared__ float sRow[16];
  __shared__ float sPart[4][256];    // split-K partials

  const int tid = threadIdx.x, wave = tid >> 5, lane = tid & 31;
  const int qt = blockIdx.x & 63;
  const int h  = (blockIdx.x >> 6) & 15;
  const int b  = blockIdx.x >> 10;
  const int qBase = qt * 16;

  const _Float16* qp = qh  + ((size_t)(b * NH + h) * SEQ + qBase) * DK;
  const _Float16* kp = kh  + (size_t)(b * NH + h) * SEQ * DK;   // [kv][64]
  const _Float16* vp = vhT + (size_t)(b * NH + h) * DK * SEQ;   // [d][1024]

  const v16h a0 = frag_a_f16(qp, DK, lane);
  const v16h a1 = frag_a_f16(qp + 32, DK, lane);

  // ---- Phase 1: scores (kh rows are exactly the B^T layout for Q.K^T).
  for (int ct = wave; ct < 64; ct += 8) {
    v8f acc = {};
    const v16h b0 = frag_b_t(kp + (size_t)ct * 16 * DK, DK, lane);
    const v16h b1 = frag_b_t(kp + (size_t)ct * 16 * DK + 32, DK, lane);
    acc = wmma32(a0, b0, acc);
    acc = wmma32(a1, b1, acc);
    const int colIn = lane & 15, hi = lane >> 4;
    const int col = ct * 16 + colIn;
#pragma unroll
    for (int r = 0; r < 8; ++r) {
      const int rr = r + 8 * hi;
      float s = acc[r] * 0.125f;  // 1/sqrt(64)
      if (mask[((size_t)b * SEQ + qBase + rr) * SEQ + col]) s = -__builtin_inff();
      sS[rr * 1024 + col] = s;
    }
  }
  __syncthreads();

  // ---- Phase 2: softmax (16 threads per row, stride-16 columns).
  const int row = tid >> 4, sub = tid & 15;
  float m = -__builtin_inff();
  for (int i = 0; i < 64; ++i) m = fmaxf(m, sS[row * 1024 + sub + i * 16]);
  sRed[row * 16 + sub] = m;
  __syncthreads();
  if (sub == 0) {
    float mm = -__builtin_inff();
    for (int i = 0; i < 16; ++i) mm = fmaxf(mm, sRed[row * 16 + i]);
    sRow[row] = mm;
  }
  __syncthreads();
  const float rmax = sRow[row];
  float sum = 0.f;
  for (int i = 0; i < 64; ++i) {
    const int c = sub + i * 16;
    const float e = __expf(sS[row * 1024 + c] - rmax);
    sS[row * 1024 + c] = e;
    sum += e;
  }
  __syncthreads();
  sRed[row * 16 + sub] = sum;
  __syncthreads();
  if (sub == 0) {
    float s = 0.f;
    for (int i = 0; i < 16; ++i) s += sRed[row * 16 + i];
    sRow[row] = s;
  }
  __syncthreads();
  const float inv = 1.0f / sRow[row];
  float* arow = attn + ((size_t)(h * BATCH + b) * SEQ + qBase + row) * SEQ;
  for (int i = 0; i < 64; ++i) {
    const int c = sub + i * 16;
    const float p = sS[row * 1024 + c] * inv;
    sS[row * 1024 + c] = p;   // keep normalized P for P*V
    arow[c] = p;              // fp32 attention output (dominant HBM write)
  }
  __syncthreads();

  // ---- Phase 3: O(16x64) = P(16x1024) @ V(1024x64), split-K over wave pairs.
  const int ctile = wave & 3;
  const int khalf = wave >> 2;
  v8f acc = {};
  for (int chunk = khalf * 16; chunk < khalf * 16 + 16; ++chunk) {
    const int kb = chunk * 32;
    const int ar = lane & 15;
    const int ko = kb + ((lane >> 4) ? 8 : 0);
    v16h af;
#pragma unroll
    for (int j = 0; j < 8; ++j) {
      af[j]     = (_Float16)sS[ar * 1024 + ko + j];
      af[j + 8] = (_Float16)sS[ar * 1024 + ko + 16 + j];
    }
    const v16h bf = frag_b_t(vp + (size_t)ctile * 16 * SEQ + kb, SEQ, lane);
    acc = wmma32(af, bf, acc);
  }
  if (khalf == 1) {
#pragma unroll
    for (int r = 0; r < 8; ++r) sPart[ctile][r * 32 + lane] = acc[r];
  }
  __syncthreads();
  if (khalf == 0) {
    const int colIn = lane & 15, hi = lane >> 4;
#pragma unroll
    for (int r = 0; r < 8; ++r) {
      const float v = acc[r] + sPart[ctile][r * 32 + lane];
      const int d = ctile * 16 + colIn;
      const int q = qBase + r + 8 * hi;
      headout[((size_t)(b * SEQ + q) << 10) + h * DK + d] = (_Float16)v;
    }
  }
}

// ---------------------------------------------------------------------------
// Row LayerNorm over d_model=1024 (bias/residual already folded into x).
// ---------------------------------------------------------------------------
__global__ __launch_bounds__(256) void ln_kernel(
    const float* __restrict__ x, const float* __restrict__ g,
    const float* __restrict__ bta, float* __restrict__ out) {
  __shared__ float red[256];
  const int row = blockIdx.x, tid = threadIdx.x;
  float v[4];
  float s = 0.f;
#pragma unroll
  for (int i = 0; i < 4; ++i) {
    v[i] = x[(size_t)row * 1024 + tid + i * 256];
    s += v[i];
  }
  red[tid] = s;
  __syncthreads();
  for (int off = 128; off > 0; off >>= 1) {
    if (tid < off) red[tid] += red[tid + off];
    __syncthreads();
  }
  const float mean = red[0] * (1.0f / 1024.0f);
  __syncthreads();
  s = 0.f;
#pragma unroll
  for (int i = 0; i < 4; ++i) {
    const float d = v[i] - mean;
    s += d * d;
  }
  red[tid] = s;
  __syncthreads();
  for (int off = 128; off > 0; off >>= 1) {
    if (tid < off) red[tid] += red[tid + off];
    __syncthreads();
  }
  const float rstd = rsqrtf(red[0] * (1.0f / 1024.0f) + 1e-5f);
#pragma unroll
  for (int i = 0; i < 4; ++i) {
    const int c = tid + i * 256;
    out[(size_t)row * 1024 + c] = (v[i] - mean) * rstd * g[c] + bta[c];
  }
}

// ---------------------------------------------------------------------------
extern "C" void kernel_launch(void* const* d_in, const int* in_sizes, int n_in,
                              void* d_out, int out_size, void* d_ws, size_t ws_size,
                              hipStream_t stream) {
  (void)in_sizes; (void)n_in; (void)out_size; (void)ws_size;

  const float* q    = (const float*)d_in[0];
  const float* k    = (const float*)d_in[1];
  const float* v    = (const float*)d_in[2];
  const unsigned char* mask = (const unsigned char*)d_in[3];
  const float* w_q  = (const float*)d_in[4];
  const float* b_q  = (const float*)d_in[5];
  const float* w_k  = (const float*)d_in[6];
  const float* b_k  = (const float*)d_in[7];
  const float* w_v  = (const float*)d_in[8];
  const float* b_v  = (const float*)d_in[9];
  const float* w_o  = (const float*)d_in[10];
  const float* b_o  = (const float*)d_in[11];
  const float* ln_g = (const float*)d_in[12];
  const float* ln_b = (const float*)d_in[13];

  // Workspace (halves): qh | kh | vhT | headout | xq | xk | xv | 4x Wt | ...
  // proj (fp32, 32MB) overlays xq+xk, which are dead by the final GEMM.
  const size_t HSZ = (size_t)BATCH * NH * SEQ * DK;  // 8M halves
  const size_t WSZ = (size_t)DM * DM;                // 1M halves
  _Float16* qh      = (_Float16*)d_ws;
  _Float16* kh      = qh + HSZ;
  _Float16* vhT     = kh + HSZ;
  _Float16* headout = vhT + HSZ;
  _Float16* xq      = headout + HSZ;
  _Float16* xk      = xq + HSZ;
  _Float16* xv      = xk + HSZ;
  _Float16* wqT     = xv + HSZ;
  _Float16* wkT     = wqT + WSZ;
  _Float16* wvT     = wkT + WSZ;
  _Float16* woT     = wvT + WSZ;
  float*    proj    = (float*)xq;   // overlay (safe: xq/xk dead by then)

  float* out  = (float*)d_out;
  float* attn = out + (size_t)BATCH * SEQ * DM;

  const dim3 blk(256);
  const int cvtGrid  = (int)(HSZ / 1024);                 // 8192
  const int gemmGrid = (BATCH * SEQ / 128) * (DM / 64);   // 1024
  const int trGrid   = (DM / 32) * (DM / 32);             // 1024

  convert_f16_kernel<<<cvtGrid, blk, 0, stream>>>(q, xq);
  convert_f16_kernel<<<cvtGrid, blk, 0, stream>>>(k, xk);
  convert_f16_kernel<<<cvtGrid, blk, 0, stream>>>(v, xv);
  transpose_f16_kernel<<<trGrid, blk, 0, stream>>>(w_q, wqT);
  transpose_f16_kernel<<<trGrid, blk, 0, stream>>>(w_k, wkT);
  transpose_f16_kernel<<<trGrid, blk, 0, stream>>>(w_v, wvT);
  transpose_f16_kernel<<<trGrid, blk, 0, stream>>>(w_o, woT);

  gemm_kernel<0><<<gemmGrid, blk, 0, stream>>>(xq, wqT, b_q, nullptr, qh);
  gemm_kernel<0><<<gemmGrid, blk, 0, stream>>>(xk, wkT, b_k, nullptr, kh);
  gemm_kernel<1><<<gemmGrid, blk, 0, stream>>>(xv, wvT, b_v, nullptr, vhT);

  attn_kernel<<<BATCH * NH * (SEQ / 16), blk, 0, stream>>>(qh, kh, vhT, mask,
                                                           attn, headout);

  gemm_kernel<2><<<gemmGrid, blk, 0, stream>>>(headout, woT, b_o, q, proj);
  ln_kernel<<<BATCH * SEQ, blk, 0, stream>>>(proj, ln_g, ln_b, out);
}
